// attention_block_84370337563058
// MI455X (gfx1250) — matrix operand order repaired and checked by reading, not compiled, not run on hardware
//
#include <hip/hip_runtime.h>
#include <hip/hip_bf16.h>
#include <stdint.h>

// ---------------------------------------------------------------------------
// 3D windowed attention block on gfx1250 (wave32, WMMA bf16 -> f32 accum).
//
// Shapes: C=128, NH=8, hd=16, windows 4x4x4 (L=64), grid 30x62x126 padded to
// 32x64x128 -> nW=4096 windows, 262144 tokens.
// ---------------------------------------------------------------------------

#define C_DIM 128
#define D0v 30
#define H0v 62
#define W0v 126
#define NWIN 4096
#define NTOK (NWIN * 64)

typedef __attribute__((ext_vector_type(16))) __bf16 v16bf;
typedef __attribute__((ext_vector_type(8)))  float  v8f;
typedef unsigned short u16b;

union Frag {
    v16bf v;
    uint4 q[2];
};

__device__ __forceinline__ u16b f2bf(float f) {
    union { float f; uint32_t u; } c; c.f = f;
    uint32_t r = c.u + 0x7FFFu + ((c.u >> 16) & 1u);   // round-nearest-even
    return (u16b)(r >> 16);
}

// ---------------------------------------------------------------------------
// Kernel 1: fp32 -> bf16 weight conversion (qkv_w 384x128, conv_w 128x128)
// ---------------------------------------------------------------------------
__global__ void k_convert_w(const float* __restrict__ qkv_w,
                            const float* __restrict__ conv_w,
                            u16b* __restrict__ qkv_wb,
                            u16b* __restrict__ conv_wb) {
    int i = blockIdx.x * blockDim.x + threadIdx.x;
    if (i < 3 * C_DIM * C_DIM) qkv_wb[i] = f2bf(qkv_w[i]);
    if (i < C_DIM * C_DIM)     conv_wb[i] = f2bf(conv_w[i]);
}

// ---------------------------------------------------------------------------
// Kernel 2: pad + window-gather x (C,30,62,126) fp32 -> xw (token, C) bf16.
// Writes fully coalesced; the channel-strided reads are absorbed by the
// 192MB L2 (x is 120MB, fits entirely).
// ---------------------------------------------------------------------------
__global__ void k_gather(const float* __restrict__ x, u16b* __restrict__ xw) {
    int idx = blockIdx.x * 256 + threadIdx.x;      // token*128 + c
    int c = idx & 127;
    int t = idx >> 7;
    int win = t >> 6, l = t & 63;
    int wd = win >> 9, wh = (win >> 5) & 15, ww = win & 31;
    int dd = l >> 4,  hh = (l >> 2) & 3,     wc = l & 3;
    int d = wd * 4 + dd, h = wh * 4 + hh, w = ww * 4 + wc;
    float v = 0.0f;
    if (d < D0v && h < H0v && w < W0v)
        v = x[((c * D0v + d) * H0v + h) * W0v + w];
    xw[idx] = f2bf(v);
}

// ---------------------------------------------------------------------------
// Kernel 3: QKV GEMM. One wave per 16x16 output tile, K=128 (4 WMMA steps).
// Col tile ct in [0,24): t3=ct/8 selects q/k/v, head=ct&7.  q,k stored
// (win,head,l,e) row-major; v stored transposed (win,head,e,l) so the
// attention B-fragment loads are contiguous.  Destination selection is
// hoisted out of the store loop (base pointer + stride), so the epilogue
// is a branch-free run of 8 stores.
// ---------------------------------------------------------------------------
__global__ void __launch_bounds__(256)
k_qkv(const u16b* __restrict__ xw, const u16b* __restrict__ qkv_wb,
      const float* __restrict__ qkv_b,
      u16b* __restrict__ qo, u16b* __restrict__ ko, u16b* __restrict__ vT) {
    int wid  = threadIdx.x >> 5;
    int lane = threadIdx.x & 31;
    int hi = lane >> 4, ln = lane & 15;
    int gw = blockIdx.x * 8 + wid;
    int ct = gw % 24;            // column tile (24 * 16 = 384 channels)
    int rt = gw / 24;            // token tile  (16384 * 16 tokens)
    int tb = rt * 16;

    const u16b* arow = xw     + (size_t)(tb + ln) * C_DIM;        // A: token rows
    const u16b* brow = qkv_wb + (size_t)(ct * 16 + ln) * C_DIM;   // Bt: weight rows

    v8f acc = {};
    for (int ks = 0; ks < 4; ++ks) {
        Frag a, b;
        a.q[0] = *(const uint4*)(arow + ks * 32 + hi * 8);        // K {0-7 | 8-15}
        a.q[1] = *(const uint4*)(arow + ks * 32 + hi * 8 + 16);   // K {16-23 | 24-31}
        b.q[0] = *(const uint4*)(brow + ks * 32 + hi * 8);        // K {0-7 | 8-15}
        b.q[1] = *(const uint4*)(brow + ks * 32 + hi * 8 + 16);   // K {16-23 | 24-31}
        acc = __builtin_amdgcn_wmma_f32_16x16x32_bf16(
            false, a.v, false, b.v, (short)0, acc, false, false);
    }

    float bias = qkv_b[ct * 16 + ln];      // global channel 0..383
    int t3 = ct >> 3, head = ct & 7;
    int win = tb >> 6, lbase = (tb & 63) + 8 * hi;

    // hoisted 3-way destination select: per-lane base + per-r stride
    u16b* p0;
    int stride;
    if (t3 == 2) {     // vT[(win*8+head)*16 + e][ltok], ltok advances by 1 per r
        p0 = vT + ((size_t)(win * 8 + head) * 16 + ln) * 64 + lbase;
        stride = 1;
    } else {           // q/k[(win*8+head)*64 + ltok][e], ltok advances by 1 per r
        p0 = (t3 == 0 ? qo : ko) + ((size_t)(win * 8 + head) * 64 + lbase) * 16 + ln;
        stride = 16;
    }
    for (int r = 0; r < 8; ++r)            // C-frag: M = r + 8*(lane/16)
        p0[r * stride] = f2bf(acc[r] + bias);
}

// ---------------------------------------------------------------------------
// Kernel 4: attention, 1 block = 1 (window, head), 4 waves x 16 rows.
// S = q k^T (K=16 zero-padded to 32), mask + scale, softmax via half-wave
// shuffles, P through LDS (C-layout -> A-layout), O = P V (K=64, 2 WMMAs).
// ---------------------------------------------------------------------------
__global__ void __launch_bounds__(128)
k_attn(const u16b* __restrict__ qi, const u16b* __restrict__ ki,
       const u16b* __restrict__ vT, u16b* __restrict__ attn_out) {
    __shared__ __align__(16) u16b plds[4 * 16 * 64];   // P, 16 rows per wave

    int wid  = threadIdx.x >> 5;            // row block 0..3
    int lane = threadIdx.x & 31;
    int hi = lane >> 4, ln = lane & 15;
    int wh_ = blockIdx.x;                   // win*8 + head
    int win = wh_ >> 3, head = wh_ & 7;
    int wd = win >> 9, whh = (win >> 5) & 15, www = win & 31;

    auto padflag = [&](int t) -> bool {
        int dd = t >> 4, hh2 = (t >> 2) & 3, wc = t & 3;
        return (wd == 7 && dd >= 2) | (whh == 15 && hh2 >= 2) | (www == 31 && wc >= 2);
    };

    const u16b* qb = qi + (size_t)wh_ * 64 * 16;
    const u16b* kb = ki + (size_t)wh_ * 64 * 16;
    const u16b* vb = vT + (size_t)wh_ * 16 * 64;

    // A fragment: 16 q rows, K=0..15 real, K=16..31 zero
    Frag aq;
    aq.q[0] = *(const uint4*)(qb + (wid * 16 + ln) * 16 + hi * 8);
    aq.q[1] = make_uint4(0, 0, 0, 0);

    // S = q k^T : 4 key tiles
    v8f s[4];
    for (int mb = 0; mb < 4; ++mb) {
        Frag bk;
        const u16b* kr = kb + (mb * 16 + ln) * 16 + hi * 8;
        bk.q[0] = *(const uint4*)(kr);
        bk.q[1] = make_uint4(0, 0, 0, 0);
        v8f z = {};
        s[mb] = __builtin_amdgcn_wmma_f32_16x16x32_bf16(
            false, aq.v, false, bk.v, (short)0, z, false, false);
    }

    // scale + pad mask + row max
    const float scale = 0.25f;               // hd^-0.5
    bool fm[4];
    for (int mb = 0; mb < 4; ++mb) fm[mb] = padflag(mb * 16 + ln);
    float rmax[8], rsum[8];
    for (int r = 0; r < 8; ++r) {
        bool fl = padflag(wid * 16 + r + 8 * hi);
        float mx = -3.0e38f;
        for (int mb = 0; mb < 4; ++mb) {
            float v = s[mb][r] * scale + ((fl != fm[mb]) ? -1000.0f : 0.0f);
            s[mb][r] = v;
            mx = fmaxf(mx, v);
        }
        for (int m = 1; m < 16; m <<= 1) mx = fmaxf(mx, __shfl_xor(mx, m, 32));
        rmax[r] = mx;
    }
    for (int r = 0; r < 8; ++r) {
        float sum = 0.0f;
        for (int mb = 0; mb < 4; ++mb) {
            float p = __expf(s[mb][r] - rmax[r]);
            s[mb][r] = p;
            sum += p;
        }
        for (int m = 1; m < 16; m <<= 1) sum += __shfl_xor(sum, m, 32);
        rsum[r] = 1.0f / sum;
    }

    // P -> LDS (bf16), C-frag layout to row-major 16x64 per wave
    for (int mb = 0; mb < 4; ++mb)
        for (int r = 0; r < 8; ++r)
            plds[(wid * 16 + r + 8 * hi) * 64 + mb * 16 + ln] = f2bf(s[mb][r] * rsum[r]);

    // O = P @ V, K = 64 in two WMMA steps (same-wave LDS dep -> s_wait_dscnt)
    v8f o = {};
    for (int kc = 0; kc < 2; ++kc) {
        Frag ap, bv2;
        const u16b* pr = plds + (wid * 16 + ln) * 64 + kc * 32 + hi * 8;
        ap.q[0] = *(const uint4*)(pr);
        ap.q[1] = *(const uint4*)(pr + 16);
        const u16b* vr = vb + ln * 64 + kc * 32 + hi * 8;    // vT: contiguous in m
        bv2.q[0] = *(const uint4*)(vr);
        bv2.q[1] = *(const uint4*)(vr + 16);
        o = __builtin_amdgcn_wmma_f32_16x16x32_bf16(
            false, ap.v, false, bv2.v, (short)0, o, false, false);
    }

    // store (token, C) row-major bf16 (aliases dead xw buffer)
    u16b* op = attn_out + ((size_t)(win * 64 + wid * 16 + 8 * hi)) * C_DIM
             + head * 16 + ln;
    for (int r = 0; r < 8; ++r)
        op[(size_t)r * C_DIM] = f2bf(o[r]);
}

// ---------------------------------------------------------------------------
// Kernel 5: 1x1x1 conv as z = conv_w @ y^T  (M=out channel, N=token) so the
// cropped fp32 stores are token-contiguous. One wave per 16x16 tile, K=128.
// Crop check hoisted around the whole store loop (single exec-branch).
// ---------------------------------------------------------------------------
__global__ void __launch_bounds__(256)
k_conv(const u16b* __restrict__ y, const u16b* __restrict__ conv_wb,
       const float* __restrict__ conv_b, float* __restrict__ out) {
    int wid  = threadIdx.x >> 5;
    int lane = threadIdx.x & 31;
    int hi = lane >> 4, ln = lane & 15;
    int gw = blockIdx.x * 8 + wid;
    int ot = gw & 7;            // out-channel tile 0..7
    int nt = gw >> 3;           // token tile 0..16383
    int tb = nt * 16;

    const u16b* arow = conv_wb + (size_t)(ot * 16 + ln) * C_DIM;
    const u16b* brow = y       + (size_t)(tb + ln) * C_DIM;

    v8f acc = {};
    for (int ks = 0; ks < 4; ++ks) {
        Frag a, b;
        a.q[0] = *(const uint4*)(arow + ks * 32 + hi * 8);
        a.q[1] = *(const uint4*)(arow + ks * 32 + hi * 8 + 16);
        b.q[0] = *(const uint4*)(brow + ks * 32 + hi * 8);
        b.q[1] = *(const uint4*)(brow + ks * 32 + hi * 8 + 16);
        acc = __builtin_amdgcn_wmma_f32_16x16x32_bf16(
            false, a.v, false, b.v, (short)0, acc, false, false);
    }

    // map this lane's token (N = ln) to spatial coords, crop pad
    int win = tb >> 6;
    int wd = win >> 9, wh = (win >> 5) & 15, ww = win & 31;
    int ltok = (tb & 63) + ln;
    int dd = ltok >> 4, hh = (ltok >> 2) & 3, wc = ltok & 3;
    int d = wd * 4 + dd, h = wh * 4 + hh, w = ww * 4 + wc;

    if ((d < D0v) && (h < H0v) && (w < W0v)) {
        int ob = ot * 16 + 8 * hi;                       // C-frag row = out channel
        float* po = out + ((size_t)(ob * D0v + d) * H0v + h) * W0v + w;
        const size_t ostride = (size_t)D0v * H0v * W0v;  // +1 out channel
        for (int r = 0; r < 8; ++r)
            po[(size_t)r * ostride] = acc[r] + conv_b[ob + r];
    }
}

// ---------------------------------------------------------------------------
extern "C" void kernel_launch(void* const* d_in, const int* in_sizes, int n_in,
                              void* d_out, int out_size, void* d_ws, size_t ws_size,
                              hipStream_t stream) {
    const float* x      = (const float*)d_in[0];
    const float* qkv_w  = (const float*)d_in[1];
    const float* qkv_b  = (const float*)d_in[2];
    const float* conv_w = (const float*)d_in[3];
    const float* conv_b = (const float*)d_in[4];
    float* out = (float*)d_out;

    // workspace layout (bytes); total ~269 MB
    char* ws = (char*)d_ws;
    const size_t SZ = (size_t)NTOK * C_DIM * 2;   // 67,108,864 per bf16 tensor
    u16b* xw      = (u16b*)(ws);                  // xw, later reused as attn_out
    u16b* qb      = (u16b*)(ws + SZ);
    u16b* kb      = (u16b*)(ws + 2 * SZ);
    u16b* vT      = (u16b*)(ws + 3 * SZ);
    u16b* qkv_wb  = (u16b*)(ws + 4 * SZ);
    u16b* conv_wb = (u16b*)(ws + 4 * SZ + 3 * C_DIM * C_DIM * 2);

    k_convert_w<<<256, 256, 0, stream>>>(qkv_w, conv_w, qkv_wb, conv_wb);
    k_gather<<<(NTOK * C_DIM) / 256, 256, 0, stream>>>(x, xw);
    k_qkv<<<(16384 * 24) / 8, 256, 0, stream>>>(xw, qkv_wb, qkv_b, qb, kb, vT);
    k_attn<<<NWIN * 8, 128, 0, stream>>>(qb, kb, vT, xw);   // attn_out aliases xw
    k_conv<<<(16384 * 8) / 8, 256, 0, stream>>>(xw, conv_wb, conv_b, out);
}
